// ModalUncertaintyEstimator_40132174414098
// MI455X (gfx1250) — matrix operand order repaired and checked by reading, not compiled, not run on hardware
//
#include <hip/hip_runtime.h>
#include <hip/hip_bf16.h>
#include <math.h>

// ---------------------------------------------------------------------------
// ModalUncertaintyEstimator for MI455X (gfx1250, wave32, WMMA bf16)
// B=64, C=8, H=256, M=32  -> 512 images, 1024 modes each
// ---------------------------------------------------------------------------

typedef __attribute__((ext_vector_type(16))) __bf16 v16bf;
typedef __attribute__((ext_vector_type(8)))  __bf16 v8bf;
typedef __attribute__((ext_vector_type(8)))  float  v8f;

#define NIMG   512
#define NMODE  1024
#define EPSF   1e-8f

// output section offsets (floats), reference tuple order
#define OUT_EN    0u
#define OUT_UNC   524288u
#define OUT_FRAC  1048576u
#define OUT_WI    1572864u
#define OUT_USPEC 2097152u
#define OUT_ESPEC 2098176u
#define OUT_ERR   2099200u
#define OUT_CAL   2623488u

// dynamic LDS layout (bytes)
#define SM_X    0        // bf16 [64][256]  image chunk          32768
#define SM_YTR  32768    // bf16 [32][256]  Y real, transposed   16384
#define SM_YTI  49152    // bf16 [32][256]  Y imag, transposed   16384
#define SM_CT   65536    // bf16 [32][256]  cos twiddle (k-major)16384
#define SM_ST   81920    // bf16 [32][256]  sin twiddle          16384
#define SM_NST  98304    // bf16 [32][256]  -sin twiddle         16384
#define SM_ZPR  114688   // f32  [1024]  pred modes real          4096
#define SM_ZPI  118784   // f32  [1024]  pred modes imag          4096
#define SM_ZGR  122880   // f32  [1024]  gt modes real            4096
#define SM_ZGI  126976   // f32  [1024]  gt modes imag            4096
#define SM_W2T  131072   // bf16 [32][64] W2 transposed           4096
#define SM_H1   135168   // bf16 [8][16][64] per-wave h1 tiles   16384
#define SM_U    151552   // f32  [1024]  uncertainties            4096
#define SM_W1   155648   // f32  [128]                              512
#define SM_B1   156160   // f32  [64]                               256
#define SM_W3   156416   // f32  [32]                               128
#define SM_B2   156544   // f32  [32]                               128
#define SM_RED  156672   // f32  [256]                             1024
#define SM_B3   157696   // f32  [1]
#define SMEM_BYTES 157760

static __device__ __forceinline__ v8f wmma_bf16(v16bf a, v16bf b, v8f c) {
  // D = A(16x32 bf16) * B(32x16 bf16) + C(16x16 f32)
  return __builtin_amdgcn_wmma_f32_16x16x32_bf16(
      /*neg_a=*/false, a, /*neg_b=*/false, b,
      /*c_mod=*/(short)0, c, /*reuse_a=*/false, /*reuse_b=*/false);
}

// A fragment (16x32, MxK), source row-major src[m][k] with row stride ldk.
// ISA layout: lanes 0-15: K 0..7 & 16..23 ; lanes 16-31: K 8..15 & 24..31.
static __device__ __forceinline__ v16bf load_A(const __bf16* base, int ldk) {
  const int lane = threadIdx.x & 31;
  const __bf16* p = base + (lane & 15) * ldk + (lane >> 4) * 8;
  v8bf lo = *(const v8bf*)(p);
  v8bf hi = *(const v8bf*)(p + 16);
  return __builtin_shufflevector(lo, hi, 0,1,2,3,4,5,6,7,8,9,10,11,12,13,14,15);
}

// B fragment (32x16, KxN), source stored TRANSPOSED: srcT[n][k], row stride ldk.
// Layout: lanes 0-15: col n=lane, K 0..7 & 16..23 ; lanes 16-31: col n=lane-16, K 8..15 & 24..31.
static __device__ __forceinline__ v16bf load_B(const __bf16* baseT, int ldk) {
  const int lane = threadIdx.x & 31;
  const __bf16* p = baseT + (lane & 15) * ldk + (lane >> 4) * 8;
  v8bf lo = *(const v8bf*)(p);
  v8bf hi = *(const v8bf*)(p + 16);
  return __builtin_shufflevector(lo, hi, 0,1,2,3,4,5,6,7,8,9,10,11,12,13,14,15);
}

__global__ __launch_bounds__(256) void modal_main(
    const float* __restrict__ pred, const float* __restrict__ gt,
    const float* __restrict__ W1, const float* __restrict__ b1,
    const float* __restrict__ W2, const float* __restrict__ b2,
    const float* __restrict__ W3, const float* __restrict__ b3,
    float* __restrict__ out)
{
  extern __shared__ char smem[];
  __bf16* X    = (__bf16*)(smem + SM_X);
  __bf16* YtR  = (__bf16*)(smem + SM_YTR);
  __bf16* YtI  = (__bf16*)(smem + SM_YTI);
  __bf16* CT   = (__bf16*)(smem + SM_CT);
  __bf16* ST   = (__bf16*)(smem + SM_ST);
  __bf16* NST  = (__bf16*)(smem + SM_NST);
  float*  Zpr  = (float*)(smem + SM_ZPR);
  float*  Zpi  = (float*)(smem + SM_ZPI);
  float*  Zgr  = (float*)(smem + SM_ZGR);
  float*  Zgi  = (float*)(smem + SM_ZGI);
  __bf16* W2t  = (__bf16*)(smem + SM_W2T);
  __bf16* H1   = (__bf16*)(smem + SM_H1);
  float*  Ubuf = (float*)(smem + SM_U);
  float*  W1s  = (float*)(smem + SM_W1);
  float*  b1s  = (float*)(smem + SM_B1);
  float*  W3s  = (float*)(smem + SM_W3);
  float*  b2s  = (float*)(smem + SM_B2);
  float*  red  = (float*)(smem + SM_RED);
  float*  b3s  = (float*)(smem + SM_B3);

  const int tid  = threadIdx.x;
  const int lane = tid & 31;
  const int w    = tid >> 5;          // wave id, 0..7
  const int img  = blockIdx.x;        // (b,c) image index, 0..511

  // ---- Phase 0: twiddles + weights into LDS -------------------------------
  // twiddle[k][n] for k in 0..31, n in 0..255 ; angle uses (k*n) mod 256.
  for (int i = tid; i < 8192; i += 256) {
    int kk = i >> 8, nn = i & 255;
    float th = (float)((kk * nn) & 255) * 0.024543692606170f; // 2*pi/256
    float sv, cv;
    __sincosf(th, &sv, &cv);
    CT[i]  = (__bf16)cv;
    ST[i]  = (__bf16)sv;
    NST[i] = (__bf16)(-sv);
  }
  if (tid < 128) W1s[tid] = W1[tid];        // W1 [2][64]
  if (tid < 64)  b1s[tid] = b1[tid];
  if (tid < 32)  { W3s[tid] = W3[tid]; b2s[tid] = b2[tid]; }
  if (tid == 0)  b3s[0] = b3[0];
  for (int i = tid; i < 2048; i += 256) {   // W2t[n][k] = W2[k][n]
    int n = i >> 6, k = i & 63;
    W2t[i] = (__bf16)W2[k * 32 + n];
  }
  __syncthreads();

  // ---- Phase 1: truncated 2D DFT of pred and gt via WMMA ------------------
  for (int which = 0; which < 2; ++which) {
    const float* src = (which == 0 ? pred : gt) + (size_t)img * 65536;
    float* Zr = which == 0 ? Zpr : Zgr;
    float* Zi = which == 0 ? Zpi : Zgi;

    // Stage 1: Y[n1, k2] = sum_n2 x[n1,n2] * e^{-2pi i n2 k2/256}, 64 rows/chunk
    for (int chunk = 0; chunk < 4; ++chunk) {
      if (chunk < 3)
        __builtin_prefetch((const void*)(src + (chunk + 1) * 16384 + tid * 64), 0, 1);
      const float4* s4 = (const float4*)(src + chunk * 16384);
      for (int i = tid; i < 4096; i += 256) {
        float4 v = s4[i];
        __bf16* d = X + 4 * i;
        d[0] = (__bf16)v.x; d[1] = (__bf16)v.y;
        d[2] = (__bf16)v.z; d[3] = (__bf16)v.w;
      }
      __syncthreads();
      {
        const int rt = w & 3, ct = w >> 2;       // 4 row tiles x 2 col tiles
        v8f yr = {}, yi = {};
        const __bf16* Ab = X   + rt * 16 * 256;
        const __bf16* Bc = CT  + ct * 16 * 256;
        const __bf16* Bn = NST + ct * 16 * 256;
        for (int kk = 0; kk < 8; ++kk) {         // K = 256 over n2
          v16bf a = load_A(Ab + kk * 32, 256);
          yr = wmma_bf16(a, load_B(Bc + kk * 32, 256), yr);  // Yr =  X*cos
          yi = wmma_bf16(a, load_B(Bn + kk * 32, 256), yi);  // Yi = -X*sin
        }
        const int h = lane >> 4, n = lane & 15;
        for (int r = 0; r < 8; ++r) {            // store transposed Yt[k2][n1]
          int n1 = chunk * 64 + rt * 16 + r + 8 * h;
          int k2 = ct * 16 + n;
          YtR[k2 * 256 + n1] = (__bf16)yr[r];
          YtI[k2 * 256 + n1] = (__bf16)yi[r];
        }
      }
      __syncthreads();
    }

    // Stage 2: Zr = C^T Yr + S^T Yi ; Zi = C^T Yi - S^T Yr  (contract n1=256)
    {
      const int t = w & 3, part = w >> 2;        // 4 output tiles x {re,im}
      const int rt = t & 1, ct = t >> 1;
      const __bf16* A2b = (part == 0) ? ST  : NST;
      const __bf16* B1b = (part == 0) ? YtR : YtI;
      const __bf16* B2b = (part == 0) ? YtI : YtR;
      v8f z = {};
      for (int kk = 0; kk < 8; ++kk) {
        v16bf a1 = load_A(CT  + rt * 4096 + kk * 32, 256);
        v16bf a2 = load_A(A2b + rt * 4096 + kk * 32, 256);
        z = wmma_bf16(a1, load_B(B1b + ct * 4096 + kk * 32, 256), z);
        z = wmma_bf16(a2, load_B(B2b + ct * 4096 + kk * 32, 256), z);
      }
      float* Zd = (part == 0) ? Zr : Zi;
      const int h = lane >> 4, n = lane & 15;
      for (int r = 0; r < 8; ++r) {
        int k1 = rt * 16 + r + 8 * h;
        int k2 = ct * 16 + n;
        Zd[k1 * 32 + k2] = z[r];
      }
    }
    __syncthreads();
  }

  // ---- Phase 2: energies, errors, total-energy reduction ------------------
  float partE = 0.f;
  for (int i = tid; i < NMODE; i += 256) {
    float pr = Zpr[i], pi = Zpi[i];
    float en = pr * pr + pi * pi;
    out[OUT_EN + (size_t)img * NMODE + i] = en;
    partE += en;
    float dr = pr - Zgr[i], di = pi - Zgi[i];
    out[OUT_ERR + (size_t)img * NMODE + i] = dr * dr + di * di;
  }
  red[tid] = partE;
  __syncthreads();
  for (int s = 128; s > 0; s >>= 1) {
    if (tid < s) red[tid] += red[tid + s];
    __syncthreads();
  }
  const float tot = red[0];
  const float b3v = b3s[0];

  // ---- Phase 3: per-mode MLP (2->64->32->1), layer 2 via WMMA -------------
  {
    __bf16* h1w = H1 + w * (16 * 64);           // private per-wave tile
    const int n = lane & 15, h = lane >> 4;
    const v16bf b00 = load_B(W2t, 64);            // K 0..31,  N 0..15
    const v16bf b01 = load_B(W2t + 32, 64);       // K 32..63, N 0..15
    const v16bf b10 = load_B(W2t + 16 * 64, 64);  // K 0..31,  N 16..31
    const v16bf b11 = load_B(W2t + 16 * 64 + 32, 64);
    const float w3a = W3s[n],      w3b = W3s[16 + n];
    const float b2a = b2s[n],      b2b = b2s[16 + n];

    for (int tile = w; tile < 64; tile += 8) {  // 64 tiles of 16 modes
      const int mbase = tile * 16;
      const float xr = Zpr[mbase + n];
      const float xi = Zpi[mbase + n];
      const int fb = h * 32;                    // each lane: 32 of 64 features
      for (int j = 0; j < 32; ++j) {
        int f = fb + j;
        float hv = xr * W1s[f] + xi * W1s[64 + f] + b1s[f];
        h1w[n * 64 + f] = (__bf16)fmaxf(hv, 0.f);
      }
      asm volatile("" ::: "memory");            // same-wave DS ops are in-order
      v16bf a0 = load_A(h1w, 64);
      v16bf a1 = load_A(h1w + 32, 64);
      v8f c0 = {}, c1 = {};
      c0 = wmma_bf16(a0, b00, c0);
      c0 = wmma_bf16(a1, b01, c0);
      c1 = wmma_bf16(a0, b10, c1);
      c1 = wmma_bf16(a1, b11, c1);
      // layer 3 + softplus: reduce 32 output features across 16 lanes
      for (int r = 0; r < 8; ++r) {
        float s = fmaxf(c0[r] + b2a, 0.f) * w3a + fmaxf(c1[r] + b2b, 0.f) * w3b;
        s += __shfl_xor(s, 1, 16);
        s += __shfl_xor(s, 2, 16);
        s += __shfl_xor(s, 4, 16);
        s += __shfl_xor(s, 8, 16);
        if (n == 0) {
          int m = mbase + r + 8 * h;
          float x = s + b3v;
          float u = (x > 20.f) ? x : log1pf(__expf(x));
          Ubuf[m] = u;
          out[OUT_UNC + (size_t)img * NMODE + m] = u;
        }
      }
      asm volatile("" ::: "memory");
    }
  }
  __syncthreads();

  // ---- Phase 4: fractions & weighted importance ---------------------------
  const float inv = 1.0f / (tot + EPSF);
  for (int i = tid; i < NMODE; i += 256) {
    float pr = Zpr[i], pi = Zpi[i];
    float fr = (pr * pr + pi * pi) * inv;
    out[OUT_FRAC + (size_t)img * NMODE + i] = fr;
    out[OUT_WI   + (size_t)img * NMODE + i] = fr * Ubuf[i];
  }
}

// ---------------------------------------------------------------------------
// Per-mode reduction over 512 images: spectra means + Pearson calibration.
// Deterministic tree reduction (no fp atomics).
// ---------------------------------------------------------------------------
__global__ __launch_bounds__(256) void modal_reduce(float* __restrict__ out)
{
  __shared__ float r0[256], r1[256], r2[256], r3[256], r4[256], r5[256];
  const int tid  = threadIdx.x;
  const int mode = blockIdx.x;
  const float* u  = out + OUT_UNC;
  const float* e  = out + OUT_ERR;
  const float* en = out + OUT_EN;

  float su = 0.f, se = 0.f, suu = 0.f, see = 0.f, sue = 0.f, sen = 0.f;
  for (int img = tid; img < NIMG; img += 256) {
    float uv = u[(size_t)img * NMODE + mode];
    float ev = e[(size_t)img * NMODE + mode];
    float nv = en[(size_t)img * NMODE + mode];
    su += uv; se += ev;
    suu += uv * uv; see += ev * ev; sue += uv * ev;
    sen += nv;
  }
  r0[tid] = su; r1[tid] = se; r2[tid] = suu;
  r3[tid] = see; r4[tid] = sue; r5[tid] = sen;
  __syncthreads();
  for (int s = 128; s > 0; s >>= 1) {
    if (tid < s) {
      r0[tid] += r0[tid + s]; r1[tid] += r1[tid + s];
      r2[tid] += r2[tid + s]; r3[tid] += r3[tid + s];
      r4[tid] += r4[tid + s]; r5[tid] += r5[tid + s];
    }
    __syncthreads();
  }
  if (tid == 0) {
    const float N = (float)NIMG;
    float mu = r0[0] / N, me = r1[0] / N;
    out[OUT_USPEC + mode] = mu;          // mean uncertainty spectrum
    out[OUT_ESPEC + mode] = r5[0] / N;   // mean energy spectrum
    float num = r4[0] - N * mu * me;     // sum uc*ec
    float va  = r2[0] - N * mu * mu;     // sum uc^2
    float vb  = r3[0] - N * me * me;     // sum ec^2
    float den = sqrtf(fmaxf(va, 0.f) * fmaxf(vb, 0.f));
    out[OUT_CAL + mode] = num / (den + EPSF);
  }
}

extern "C" void kernel_launch(void* const* d_in, const int* in_sizes, int n_in,
                              void* d_out, int out_size, void* d_ws, size_t ws_size,
                              hipStream_t stream) {
  const float* pred = (const float*)d_in[0];
  // d_in[1] = 'uncertainty' — unused by the reference computation
  const float* gt = (const float*)d_in[2];
  const float* W1 = (const float*)d_in[3];
  const float* b1 = (const float*)d_in[4];
  const float* W2 = (const float*)d_in[5];
  const float* b2 = (const float*)d_in[6];
  const float* W3 = (const float*)d_in[7];
  const float* b3 = (const float*)d_in[8];
  float* out = (float*)d_out;

  (void)in_sizes; (void)n_in; (void)out_size; (void)d_ws; (void)ws_size;

  // allow >64KB dynamic LDS (CDNA5 WGP has 320KB)
  hipFuncSetAttribute((const void*)modal_main,
                      hipFuncAttributeMaxDynamicSharedMemorySize, SMEM_BYTES);

  modal_main<<<dim3(NIMG), dim3(256), SMEM_BYTES, stream>>>(
      pred, gt, W1, b1, W2, b2, W3, b3, out);
  modal_reduce<<<dim3(NMODE), dim3(256), 0, stream>>>(out);
}
